// FlashCrossAttentionLayer_31353261261323
// MI455X (gfx1250) — matrix operand order repaired and checked by reading, not compiled, not run on hardware
//
#include <hip/hip_runtime.h>
#include <hip/hip_bf16.h>
#include <cstdint>
#include <cstddef>

// ---------------------------------------------------------------------------
// FlashCrossAttentionLayer for MI455X (gfx1250, wave32, WMMA, TDM).
// Pipeline: cvt weights -> LN(q),LN(kv) -> GEMM qh -> GEMM kvp -> pack K/V^T
//           -> flash attention (bf16 WMMA, fp32 softmax, TDM-staged K tiles)
//           -> GEMM out (+resid).
// Matrix math: v_wmma_f32_16x16x32_bf16 with fp32 accumulation.
// Softmax row-sums on the matrix unit (P @ ones accumulator); row-max via
// ds_swizzle XOR patterns. K tiles DMA'd to LDS with tensor_load_to_lds.
// ---------------------------------------------------------------------------

typedef __bf16 bf16_t;
typedef __attribute__((ext_vector_type(16))) __bf16 v16bf;
typedef __attribute__((ext_vector_type(8)))  float  v8f;
typedef __attribute__((ext_vector_type(4)))  float  v4f;
typedef __attribute__((ext_vector_type(4)))  unsigned int v4u;
typedef __attribute__((ext_vector_type(8)))  int v8i;
typedef __attribute__((ext_vector_type(4)))  int v4i;

namespace {

constexpr int kB = 4, kLQ = 1024, kLKV = 2048, kD = 1024, kH = 16, kHD = 64;
constexpr float kEps = 1e-5f;
constexpr float kNeg = -1.0e9f;
constexpr float kScale = 0.125f;  // 1/sqrt(HD)

__device__ __forceinline__ bf16_t f2bf(float f) {
  unsigned u = __builtin_bit_cast(unsigned, f);
  u += 0x7FFFu + ((u >> 16) & 1u);            // round-to-nearest-even
  unsigned short s = (unsigned short)(u >> 16);
  return __builtin_bit_cast(bf16_t, s);
}

__device__ __forceinline__ v8f wmma_bf16(v16bf a, v16bf b, v8f c) {
  return __builtin_amdgcn_wmma_f32_16x16x32_bf16(
      /*neg_a=*/false, a, /*neg_b=*/false, b,
      /*c_mod=*/(short)0, c, /*reuse_a=*/false, /*reuse_b=*/false);
}

// max-reduce across the 16 lanes of a half-wave via ds_swizzle XOR patterns
// (offset = xor_mask<<10 | 0x1F; xor masks <= 8 never cross the 16-lane half).
__device__ __forceinline__ float max_xor16(float x) {
  x = fmaxf(x, __builtin_bit_cast(float, __builtin_amdgcn_ds_swizzle(
                   __builtin_bit_cast(int, x), 0x201F)));  // SWAPX8
  x = fmaxf(x, __builtin_bit_cast(float, __builtin_amdgcn_ds_swizzle(
                   __builtin_bit_cast(int, x), 0x101F)));  // SWAPX4
  x = fmaxf(x, __builtin_bit_cast(float, __builtin_amdgcn_ds_swizzle(
                   __builtin_bit_cast(int, x), 0x081F)));  // SWAPX2
  x = fmaxf(x, __builtin_bit_cast(float, __builtin_amdgcn_ds_swizzle(
                   __builtin_bit_cast(int, x), 0x041F)));  // SWAPX1
  return x;
}

// A-fragment, 16(M) x 32(K) bf16, row-major source with leading dim `ld`.
// ISA layout: lanes 0-15 = rows, K {0..7, 16..23}; lanes 16-31 same rows,
// K {8..15, 24..31}; two 16B loads per lane.
__device__ __forceinline__ v16bf load_fragA(const bf16_t* __restrict__ p,
                                            int row, int k, int ld) {
  const int lane = threadIdx.x & 31;
  const int r = lane & 15, hh = (lane >> 4) & 1;
  const bf16_t* base = p + (size_t)(row + r) * ld + k + hh * 8;
  v16bf f;
  ((v4f*)&f)[0] = *(const v4f*)(base);        // K + {0..7}
  ((v4f*)&f)[1] = *(const v4f*)(base + 16);   // K + {16..23}
  return f;
}

// B-fragment, 32(K) x 16(N) bf16. Column n of B comes from row (nrow+n) of a
// row-major (N,K) matrix (weights stored (N,K); V^T stored (HD,LKV)).
// ISA layout: lanes 0-15 hold K {0..7, 16..23}, lanes 16-31 hold K {8..15, 24..31};
// lane&15 = N. Two 16B loads per lane.
__device__ __forceinline__ v16bf load_fragB(const bf16_t* __restrict__ p,
                                            int nrow, int k, int ld) {
  const int lane = threadIdx.x & 31;
  const int n = lane & 15, hh = (lane >> 4) & 1;
  const bf16_t* base = p + (size_t)(nrow + n) * ld + k + hh * 8;
  v16bf f;
  ((v4f*)&f)[0] = *(const v4f*)(base);
  ((v4f*)&f)[1] = *(const v4f*)(base + 16);
  return f;
}

// ---------------------------------------------------------------------------
// Tensor Data Mover: DMA a 2D bf16 tile (tile_d0 x tile_d1 rows) from global
// memory into LDS at byte offset `lds_off`. D# per CDNA5 ISA ch.8:
//   group0: count=1 | lds_addr | global_addr(57b) | type=2
//   group1: data_size=1(2B) | tensor_dim0/1 | tile_dim0/1 | dim0_stride
//   groups 2/3: zero (tensor <= 2D)
// Tracked by TENSORcnt. Descriptor values are wave-uniform.
// ---------------------------------------------------------------------------
__device__ __forceinline__ void tdm_load_tile_2d(unsigned lds_off,
                                                 const void* gaddr,
                                                 unsigned tile_d0, unsigned tile_d1,
                                                 unsigned tensor_d0, unsigned tensor_d1,
                                                 unsigned d0_stride) {
  unsigned long long ga = (unsigned long long)gaddr;
  v4u g0;
  g0[0] = 1u;                                        // count=1, user D#
  g0[1] = lds_off;                                   // lds_addr (bytes)
  g0[2] = (unsigned)(ga & 0xFFFFFFFFu);              // global_addr[31:0]
  g0[3] = (unsigned)((ga >> 32) & 0x01FFFFFFu)       // global_addr[56:32]
          | (2u << 30);                              // type = 2 ("image")
  v8i g1 = {};
  g1[0] = (int)(1u << 16);                           // data_size=1 -> 2 bytes
  g1[1] = (int)((tensor_d0 & 0xFFFFu) << 16);        // tensor_dim0[15:0]
  g1[2] = (int)(((tensor_d0 >> 16) & 0xFFFFu) |
                ((tensor_d1 & 0xFFFFu) << 16));      // dim0[31:16] | dim1[15:0]
  g1[3] = (int)(((tensor_d1 >> 16) & 0xFFFFu) |
                ((tile_d0 & 0xFFFFu) << 16));        // dim1[31:16] | tile_dim0
  g1[4] = (int)(tile_d1 & 0xFFFFu);                  // tile_dim1 (tile_dim2=0)
  g1[5] = (int)d0_stride;                            // tensor_dim0_stride[31:0]
  v4i g2 = {};
  v4i g3 = {};
#if __clang_major__ >= 23
  v8i g4 = {};
  __builtin_amdgcn_tensor_load_to_lds(g0, g1, g2, g3, g4, 0);
#else
  __builtin_amdgcn_tensor_load_to_lds(g0, g1, g2, g3, 0);
#endif
}

// ---------------------------------------------------------------------------
// fp32 -> bf16 elementwise convert (weights).
// ---------------------------------------------------------------------------
__global__ void cvt_bf16_kernel(const float* __restrict__ in,
                                bf16_t* __restrict__ out, int n) {
  int i = blockIdx.x * blockDim.x + threadIdx.x;
  if (i < n) out[i] = f2bf(in[i]);
}

// ---------------------------------------------------------------------------
// LayerNorm: one block (256 threads) per row of length D, output bf16.
// ---------------------------------------------------------------------------
__global__ void ln_bf16_kernel(const float* __restrict__ x,
                               const float* __restrict__ w,
                               const float* __restrict__ b,
                               bf16_t* __restrict__ out, int d) {
  const int row = blockIdx.x;
  const float* xr = x + (size_t)row * d;
  float s = 0.f, ss = 0.f;
  for (int i = threadIdx.x; i < d; i += blockDim.x) {
    float v = xr[i];
    s += v; ss += v * v;
  }
#pragma unroll
  for (int off = 16; off; off >>= 1) {
    s  += __shfl_xor(s,  off, 32);
    ss += __shfl_xor(ss, off, 32);
  }
  __shared__ float red[2][8];
  const int wid = threadIdx.x >> 5, lane = threadIdx.x & 31;
  if (lane == 0) { red[0][wid] = s; red[1][wid] = ss; }
  __syncthreads();
  s = 0.f; ss = 0.f;
  const int nw = blockDim.x >> 5;
  for (int i = 0; i < nw; ++i) { s += red[0][i]; ss += red[1][i]; }
  const float mu  = s / d;
  const float var = ss / d - mu * mu;
  const float rs  = rsqrtf(var + kEps);
  bf16_t* orow = out + (size_t)row * d;
  for (int i = threadIdx.x; i < d; i += blockDim.x)
    orow[i] = f2bf((xr[i] - mu) * rs * w[i] + b[i]);
}

// ---------------------------------------------------------------------------
// GEMM: C(M,N) = A(M,K)bf16 @ W(N,K)bf16^T + bias (+ resid).
// Block = 256 threads (8 waves) on a 128x128 tile; each wave owns a 32x64
// sub-tile: 2 A-frags x 4 B-frags -> 8 WMMAs per 12 b128 loads per K-step.
// Weights/activations stream from the 192MB L2; no LDS staging needed.
// ---------------------------------------------------------------------------
template <bool OUT_BF16, bool RESID>
__global__ void gemm_bf16_kernel(const bf16_t* __restrict__ A,
                                 const bf16_t* __restrict__ W,
                                 const float* __restrict__ bias,
                                 const float* __restrict__ resid,
                                 void* __restrict__ out,
                                 int M, int N, int K) {
  const int wid = threadIdx.x >> 5;
  const int m0 = blockIdx.y * 128 + (wid >> 1) * 32;  // 4 M-slots
  const int n0 = blockIdx.x * 128 + (wid & 1) * 64;   // 2 N-slots

  v8f acc[2][4];
#pragma unroll
  for (int i = 0; i < 2; ++i)
#pragma unroll
    for (int j = 0; j < 4; ++j) acc[i][j] = (v8f){};

  for (int k = 0; k < K; k += 32) {
    if (k + 32 < K) {  // speculative prefetch of next K slab
      __builtin_prefetch(A + (size_t)m0 * K + k + 32, 0, 1);
      __builtin_prefetch(W + (size_t)n0 * K + k + 32, 0, 1);
    }
    v16bf a0 = load_fragA(A, m0,      k, K);
    v16bf a1 = load_fragA(A, m0 + 16, k, K);
    v16bf b0 = load_fragB(W, n0,      k, K);
    v16bf b1 = load_fragB(W, n0 + 16, k, K);
    v16bf b2 = load_fragB(W, n0 + 32, k, K);
    v16bf b3 = load_fragB(W, n0 + 48, k, K);
    acc[0][0] = wmma_bf16(a0, b0, acc[0][0]);
    acc[0][1] = wmma_bf16(a0, b1, acc[0][1]);
    acc[0][2] = wmma_bf16(a0, b2, acc[0][2]);
    acc[0][3] = wmma_bf16(a0, b3, acc[0][3]);
    acc[1][0] = wmma_bf16(a1, b0, acc[1][0]);
    acc[1][1] = wmma_bf16(a1, b1, acc[1][1]);
    acc[1][2] = wmma_bf16(a1, b2, acc[1][2]);
    acc[1][3] = wmma_bf16(a1, b3, acc[1][3]);
  }

  // C layout: lane<16 -> rows m..m+7 (VGPR idx), col=lane; lane>=16 -> rows +8.
  const int lane = threadIdx.x & 31;
  const int c = lane & 15, hh = (lane >> 4) & 1;
#pragma unroll
  for (int i = 0; i < 2; ++i) {
#pragma unroll
    for (int j = 0; j < 4; ++j) {
      const int rbase = m0 + i * 16 + hh * 8;
      const int col = n0 + j * 16 + c;
      const float bv = bias ? bias[col] : 0.f;
#pragma unroll
      for (int r = 0; r < 8; ++r) {
        float v = acc[i][j][r] + bv;
        size_t idx = (size_t)(rbase + r) * N + col;
        if (RESID) v += resid[idx];
        if (OUT_BF16) ((bf16_t*)out)[idx] = f2bf(v);
        else          ((float*)out)[idx]  = v;
      }
    }
  }
}

// ---------------------------------------------------------------------------
// Pack kvp (B,LKV,2,H,HD)bf16 -> Kh (B,H,LKV,HD) and V^T (B,H,HD,LKV).
// ---------------------------------------------------------------------------
__global__ void pack_kv_kernel(const bf16_t* __restrict__ kvp,
                               bf16_t* __restrict__ Kh,
                               bf16_t* __restrict__ Vt) {
  size_t i = (size_t)blockIdx.x * blockDim.x + threadIdx.x;
  if (i >= (size_t)kB * kLKV * kD) return;
  int hd = (int)(i % kHD);  size_t t = i / kHD;
  int h  = (int)(t % kH);   t /= kH;
  int kv = (int)(t % kLKV);
  int b  = (int)(t / kLKV);
  size_t rb = ((size_t)b * kLKV + kv) * (2 * kD);
  bf16_t kval = kvp[rb + h * kHD + hd];
  bf16_t vval = kvp[rb + kD + h * kHD + hd];
  Kh[(((size_t)b * kH + h) * kLKV + kv) * kHD + hd] = kval;
  Vt[(((size_t)b * kH + h) * kHD + hd) * kLKV + kv] = vval;
}

// ---------------------------------------------------------------------------
// Flash attention. Grid (LQ/64, H, B); block = 128 threads = 4 waves; each
// wave owns a 16-row Q tile, streams KV in chunks of 32 with online softmax.
// Per chunk, the wave's K tile (32 x 64 bf16, 4KB) is DMA'd into LDS by the
// Tensor Data Mover and fragments are fed to the WMMAs from LDS. P (f32
// C-layout) is staged through per-wave LDS to re-shape into a bf16 A-fragment
// for the PV WMMAs against V^T. The softmax denominator is an extra WMMA
// accumulator tile (P @ ones), rescaled by alpha like O.
// ---------------------------------------------------------------------------
__global__ void attn_kernel(const bf16_t* __restrict__ qh,
                            const bf16_t* __restrict__ Kh,
                            const bf16_t* __restrict__ Vt,
                            const unsigned char* __restrict__ mask,
                            bf16_t* __restrict__ O) {
  __shared__ __align__(16) bf16_t pbuf[4][16 * 32];   // P staging, per wave
  __shared__ __align__(16) bf16_t kbuf[4][32 * 64];   // TDM K tile, per wave

  const int wid = threadIdx.x >> 5, lane = threadIdx.x & 31;
  const int c = lane & 15, hh = (lane >> 4) & 1;
  const int b = blockIdx.z, h = blockIdx.y;
  const int q0 = blockIdx.x * 64 + wid * 16;

  const bf16_t* Q  = qh + ((size_t)b * kLQ) * kD + h * kHD;      // row q: + q*kD
  const bf16_t* Kb = Kh + (((size_t)b * kH + h) * kLKV) * kHD;   // (LKV,HD)
  const bf16_t* Vb = Vt + (((size_t)b * kH + h) * kHD) * kLKV;   // (HD,LKV)
  const unsigned char* mb = mask + (size_t)b * kLKV;

  // Low 32 bits of a flat shared-space address are the LDS byte offset.
  const unsigned kbuf_off = (unsigned)(unsigned long long)(void*)&kbuf[wid][0];
  const bf16_t* kb_lds = &kbuf[wid][0];

  const v16bf qa0 = load_fragA(Q, q0, 0,  kD);   // feat 0..31
  const v16bf qa1 = load_fragA(Q, q0, 32, kD);   // feat 32..63

  v16bf onesf;
#pragma unroll
  for (int i = 0; i < 16; ++i) onesf[i] = f2bf(1.0f);

  v8f o0 = {}, o1 = {}, o2 = {}, o3 = {}, osum = {};
  float mrow[8];
#pragma unroll
  for (int r = 0; r < 8; ++r) mrow[r] = -3.0e38f;

  for (int kv0 = 0; kv0 < kLKV; kv0 += 32) {
    if (mb[kv0]) break;  // mask is monotone (kv >= length): rest is masked

    // ---- DMA this wave's 32x64 K tile into LDS via the TDM ----
    asm volatile("s_wait_dscnt 0" ::: "memory");  // LDS reads done before overwrite
    tdm_load_tile_2d(kbuf_off, Kb + (size_t)kv0 * kHD,
                     /*tile_d0=*/kHD, /*tile_d1=*/32,
                     /*tensor_d0=*/kHD, /*tensor_d1=*/kLKV,
                     /*d0_stride=*/kHD);
    __builtin_amdgcn_s_wait_tensorcnt(0);

    // ---- S = Q @ K^T for 32 kv columns (two 16-col tiles), K from LDS ----
    v16bf kb0a = load_fragB(kb_lds, 0,  0,  kHD);
    v16bf kb0b = load_fragB(kb_lds, 0,  32, kHD);
    v16bf kb1a = load_fragB(kb_lds, 16, 0,  kHD);
    v16bf kb1b = load_fragB(kb_lds, 16, 32, kHD);
    v8f s0 = {}, s1 = {};
    s0 = wmma_bf16(qa0, kb0a, s0);
    s0 = wmma_bf16(qa1, kb0b, s0);
    s1 = wmma_bf16(qa0, kb1a, s1);
    s1 = wmma_bf16(qa1, kb1b, s1);

    const bool msk0 = mb[kv0 + c] != 0;
    const bool msk1 = mb[kv0 + 16 + c] != 0;

    // ---- online softmax (row = hh*8 + r spans the 16 lanes of a half) ----
#pragma unroll
    for (int r = 0; r < 8; ++r) {
      const float v0 = msk0 ? kNeg : s0[r] * kScale;
      const float v1 = msk1 ? kNeg : s1[r] * kScale;
      const float mm = max_xor16(fmaxf(v0, v1));
      const float mnew  = fmaxf(mrow[r], mm);
      const float alpha = __expf(mrow[r] - mnew);
      const float p0 = __expf(v0 - mnew);
      const float p1 = __expf(v1 - mnew);
      pbuf[wid][(hh * 8 + r) * 32 + c]      = f2bf(p0);
      pbuf[wid][(hh * 8 + r) * 32 + 16 + c] = f2bf(p1);
      mrow[r] = mnew;
      o0[r] = o0[r] * alpha;
      o1[r] = o1[r] * alpha;
      o2[r] = o2[r] * alpha;
      o3[r] = o3[r] * alpha;
      osum[r] = osum[r] * alpha;
    }

    // ---- reload P as bf16 A-fragment (16 x 32, row-major ld=32 in LDS) ----
    v16bf pf;
    {
      const int base = c * 32 + hh * 8;  // A-frag: lane row = lane&15
      ((v4f*)&pf)[0] = *(const v4f*)&pbuf[wid][base];
      ((v4f*)&pf)[1] = *(const v4f*)&pbuf[wid][base + 16];
    }

    // ---- O += P @ V ; denominator tile l += P @ ones ----
    o0 = wmma_bf16(pf, load_fragB(Vb, 0,  kv0, kLKV), o0);
    o1 = wmma_bf16(pf, load_fragB(Vb, 16, kv0, kLKV), o1);
    o2 = wmma_bf16(pf, load_fragB(Vb, 32, kv0, kLKV), o2);
    o3 = wmma_bf16(pf, load_fragB(Vb, 48, kv0, kLKV), o3);
    osum = wmma_bf16(pf, onesf, osum);
  }

  // ---- finalize: divide by l (every column of osum holds the row-sum) ----
#pragma unroll
  for (int r = 0; r < 8; ++r) {
    const float inv = 1.0f / osum[r];
    const int qrow = q0 + hh * 8 + r;
    bf16_t* orow = O + ((size_t)b * kLQ + qrow) * kD + h * kHD;
    orow[ 0 + c] = f2bf(o0[r] * inv);
    orow[16 + c] = f2bf(o1[r] * inv);
    orow[32 + c] = f2bf(o2[r] * inv);
    orow[48 + c] = f2bf(o3[r] * inv);
  }
}

}  // namespace

// ---------------------------------------------------------------------------
// Host launcher.
// Inputs: q, kv, mask, norm_q_w, norm_q_b, norm_kv_w, norm_kv_b,
//         Wq, bq, Wkv, bkv, Wo, bo
// ---------------------------------------------------------------------------
extern "C" void kernel_launch(void* const* d_in, const int* in_sizes, int n_in,
                              void* d_out, int out_size, void* d_ws, size_t ws_size,
                              hipStream_t stream) {
  const float* q    = (const float*)d_in[0];
  const float* kv   = (const float*)d_in[1];
  const unsigned char* mask = (const unsigned char*)d_in[2];  // numpy bool = 1B
  const float* nqw  = (const float*)d_in[3];
  const float* nqb  = (const float*)d_in[4];
  const float* nkw  = (const float*)d_in[5];
  const float* nkb  = (const float*)d_in[6];
  const float* Wq   = (const float*)d_in[7];
  const float* bq   = (const float*)d_in[8];
  const float* Wkv  = (const float*)d_in[9];
  const float* bkv  = (const float*)d_in[10];
  const float* Wo   = (const float*)d_in[11];
  const float* bo   = (const float*)d_in[12];
  float* out = (float*)d_out;

  // Workspace carve-up (bf16 buffers, 256B aligned).
  size_t off = 0;
  auto alloc_bf = [&](size_t elems) -> bf16_t* {
    bf16_t* p = (bf16_t*)((char*)d_ws + off);
    off += ((elems * sizeof(bf16_t) + 255) / 256) * 256;
    return p;
  };
  bf16_t* wq_b  = alloc_bf((size_t)kD * kD);              // 1M
  bf16_t* wkv_b = alloc_bf((size_t)2 * kD * kD);          // 2M
  bf16_t* wo_b  = alloc_bf((size_t)kD * kD);              // 1M
  bf16_t* qn    = alloc_bf((size_t)kB * kLQ * kD);        // 4M
  bf16_t* kvn   = alloc_bf((size_t)kB * kLKV * kD);       // 8M
  bf16_t* qhb   = alloc_bf((size_t)kB * kLQ * kD);        // 4M
  bf16_t* kvpb  = alloc_bf((size_t)kB * kLKV * 2 * kD);   // 16M
  bf16_t* khb   = alloc_bf((size_t)kB * kH * kLKV * kHD); // 8M
  bf16_t* vtb   = alloc_bf((size_t)kB * kH * kHD * kLKV); // 8M
  bf16_t* ob    = alloc_bf((size_t)kB * kLQ * kD);        // 4M
  (void)ws_size; (void)n_in; (void)in_sizes; (void)out_size;

  // 0) weights fp32 -> bf16
  {
    int n1 = kD * kD, n2 = 2 * kD * kD;
    cvt_bf16_kernel<<<(n1 + 255) / 256, 256, 0, stream>>>(Wq,  wq_b,  n1);
    cvt_bf16_kernel<<<(n2 + 255) / 256, 256, 0, stream>>>(Wkv, wkv_b, n2);
    cvt_bf16_kernel<<<(n1 + 255) / 256, 256, 0, stream>>>(Wo,  wo_b,  n1);
  }

  // 1) LayerNorm q / kv -> bf16
  ln_bf16_kernel<<<kB * kLQ,  256, 0, stream>>>(q,  nqw, nqb, qn,  kD);
  ln_bf16_kernel<<<kB * kLKV, 256, 0, stream>>>(kv, nkw, nkb, kvn, kD);

  // 2) qh = qn @ Wq^T + bq   (M=4096, N=1024, K=1024) -> bf16
  {
    dim3 grid(kD / 128, (kB * kLQ) / 128);
    gemm_bf16_kernel<true, false><<<grid, 256, 0, stream>>>(
        qn, wq_b, bq, nullptr, qhb, kB * kLQ, kD, kD);
  }

  // 3) kvp = kvn @ Wkv^T + bkv  (M=8192, N=2048, K=1024) -> bf16
  {
    dim3 grid((2 * kD) / 128, (kB * kLKV) / 128);
    gemm_bf16_kernel<true, false><<<grid, 256, 0, stream>>>(
        kvn, wkv_b, bkv, nullptr, kvpb, kB * kLKV, 2 * kD, kD);
  }

  // 4) pack K head-major, V transposed
  {
    size_t n = (size_t)kB * kLKV * kD;
    pack_kv_kernel<<<(unsigned)((n + 255) / 256), 256, 0, stream>>>(kvpb, khb, vtb);
  }

  // 5) flash attention -> bf16 O (B,LQ,D)
  {
    dim3 grid(kLQ / 64, kH, kB);
    attn_kernel<<<grid, 128, 0, stream>>>(qhb, khb, vtb, mask, ob);
  }

  // 6) y = q + O @ Wo^T + bo  (fp32 out, fused residual)
  {
    dim3 grid(kD / 128, (kB * kLQ) / 128);
    gemm_bf16_kernel<false, true><<<grid, 256, 0, stream>>>(
        ob, wo_b, bo, q, out, kB * kLQ, kD, kD);
  }
}
